// TSP_Decoder_28647431864779
// MI455X (gfx1250) — matrix operand order repaired and checked by reading, not compiled, not run on hardware
//
#include <hip/hip_runtime.h>
#include <hip/hip_bf16.h>

// ---------------------------------------------------------------------------
// TSP decoder (AFT attention + clipped-softmax scoring) for gfx1250 (MI455X)
// All GEMM stages run on v_wmma_f32_16x16x32_bf16 (wave32 WMMA).
// K-loops are software-pipelined (double-buffered fragments) so global b128
// loads overlap the WMMA pipe instead of s_wait_loadcnt-serializing.
// ---------------------------------------------------------------------------

typedef __attribute__((ext_vector_type(16))) __bf16 v16bf;
typedef __attribute__((ext_vector_type(8)))  __bf16 v8bf;
typedef __attribute__((ext_vector_type(8)))  float  v8f;

#define WMMA_BF16(A, B, C) \
  __builtin_amdgcn_wmma_f32_16x16x32_bf16(false, (A), false, (B), (short)0, (C), false, false)

#define V8F_ZERO {0.f, 0.f, 0.f, 0.f, 0.f, 0.f, 0.f, 0.f}

// Problem constants
#define BB   32
#define PP   100
#define PPAD 112        // 7 tiles of 16
#define NN   1000
#define NPAD 1024
#define DD   128
#define INV_SQRT_D 0.08838834764831845f
#define CLIPV 10.0f

// B fragment: lane holds column (l%16); element e -> K = (l/16)*16 + e
// (contiguous 16 K values); caller passes p = colBase + kk*32 + hi*16.
static __device__ inline v16bf ld_bfrag(const __bf16* p) {
  v8bf a = *(const v8bf*)p;
  v8bf b = *(const v8bf*)(p + 8);
  v16bf f;
#pragma unroll
  for (int e = 0; e < 8; ++e) { f[e] = a[e]; f[e + 8] = b[e]; }
  return f;
}

// A fragment (bf16 row-major row pointer): element e -> K = kOff+e
// (contiguous 16 K values); caller passes kOff = kk*32 + hi*16.
static __device__ inline v16bf ld_afrag_bf16(const __bf16* rowp, int kOff) {
  v8bf a = *(const v8bf*)(rowp + kOff);
  v8bf b = *(const v8bf*)(rowp + kOff + 8);
  v16bf f;
#pragma unroll
  for (int e = 0; e < 8; ++e) { f[e] = a[e]; f[e + 8] = b[e]; }
  return f;
}

// A fragment from f32 row (converted on the fly)
static __device__ inline v16bf ld_afrag_f32(const float* rowp, int kOff) {
  v16bf f;
#pragma unroll
  for (int e = 0; e < 8; ++e) f[e] = (__bf16)rowp[kOff + e];
#pragma unroll
  for (int e = 0; e < 8; ++e) f[e + 8] = (__bf16)rowp[kOff + 8 + e];
  return f;
}

// ---------------------------------------------------------------------------
// Prep: convert the four 128x128 weight matrices to bf16; zero N-pad rows of
// ekT / ekvT / nodes_bf16 (n in [1000,1024)) so padded K-loops contribute 0.
// ---------------------------------------------------------------------------
__global__ __launch_bounds__(256) void k_prep(
    const float* __restrict__ Wk, const float* __restrict__ Wv,
    const float* __restrict__ Wqf, const float* __restrict__ Wql,
    __bf16* __restrict__ Wkb, __bf16* __restrict__ Wvb,
    __bf16* __restrict__ Wqfb, __bf16* __restrict__ Wqlb,
    __bf16* __restrict__ ekT, __bf16* __restrict__ ekvT,
    __bf16* __restrict__ nodesb) {
  int i = blockIdx.x * 256 + threadIdx.x;
  if (i < 4 * DD * DD) {
    int m = i >> 14, idx = i & 16383;
    const float* src = (m == 0) ? Wk : (m == 1) ? Wv : (m == 2) ? Wqf : Wql;
    __bf16* dst = (m == 0) ? Wkb : (m == 1) ? Wvb : (m == 2) ? Wqfb : Wqlb;
    dst[idx] = (__bf16)src[idx];
  }
  if (i < BB * DD * (NPAD - NN)) {            // 98304 pad elements
    // ekT/ekvT layout: [b][d][n]
    int n = NN + (i % (NPAD - NN));
    int bd = i / (NPAD - NN);
    int o = bd * NPAD + n;
    ekT[o]  = (__bf16)0.f;
    ekvT[o] = (__bf16)0.f;
    // nodesb layout: [b][n][d]
    int d = i & (DD - 1);
    int j = i >> 7;
    int b = j / (NPAD - NN), nn = NN + j % (NPAD - NN);
    nodesb[(b * NPAD + nn) * DD + d] = (__bf16)0.f;
  }
}

// ---------------------------------------------------------------------------
// ebias = exp(log_scale*AFT_alpha*(-cur_dist) + ninf_mask), bf16,
// padded layout [b][112][1024] (pad rows/cols = 0 so they contribute nothing).
// ---------------------------------------------------------------------------
__global__ __launch_bounds__(256) void k_ebias(
    const float* __restrict__ cur_dist, const float* __restrict__ ninf,
    const float* __restrict__ log_scale, const float* __restrict__ aft_alpha,
    __bf16* __restrict__ ebias) {
  int i = blockIdx.x * 256 + threadIdx.x;     // over BB*PPAD*NPAD
  if (i >= BB * PPAD * NPAD) return;
  int n = i & (NPAD - 1);
  int r = (i >> 10) % PPAD;
  int b = i / (PPAD * NPAD);
  float e = 0.f;
  if (r < PP && n < NN) {
    int off = (b * PP + r) * NN + n;
    float sc = log_scale[0] * aft_alpha[0];
    e = __expf(-sc * cur_dist[off] + ninf[off]);
  }
  ebias[i] = (__bf16)e;
}

// ---------------------------------------------------------------------------
// Stage 1: k = enc@WkT, v = enc@WvT (WMMA); store exp(k) and exp(k)*v
// transposed [b][d][n] (bf16) + enc as bf16 [b][n][d].
// Grid: (B*N)/16 = 2000 blocks of 128 threads.
// ---------------------------------------------------------------------------
__global__ __launch_bounds__(128) void k_proj_kv(
    const float* __restrict__ enc,
    const __bf16* __restrict__ Wkb, const __bf16* __restrict__ Wvb,
    __bf16* __restrict__ ekT, __bf16* __restrict__ ekvT,
    __bf16* __restrict__ nodesb) {
  int tid = threadIdx.x;
  int row0 = blockIdx.x * 16;                 // flat row in [B*N]
  // convert this tile of encoded_nodes to bf16 (padded layout)
  for (int i = tid; i < 16 * DD; i += 128) {
    int r = i >> 7, d = i & (DD - 1);
    int row = row0 + r;
    int b = row / NN, n = row - b * NN;
    nodesb[(b * NPAD + n) * DD + d] = (__bf16)enc[row * DD + d];
  }
  int lane = tid & 31, w = tid >> 5, li = lane & 15, hi = lane >> 4;
  const float* arow = enc + (row0 + li) * DD;
  v16bf afr[4];
#pragma unroll
  for (int kk = 0; kk < 4; ++kk) afr[kk] = ld_afrag_f32(arow, kk * 32 + hi * 16);
#pragma unroll
  for (int t = 0; t < 2; ++t) {
    int dcol = (w * 2 + t) * 16 + li;         // output feature column
    v8f ack = V8F_ZERO, acv = V8F_ZERO;
#pragma unroll
    for (int kk = 0; kk < 4; ++kk) {
      // B(K=i, N=o) = W[o][i]: row o of W is the contiguous column
      v16bf bk = ld_bfrag(Wkb + dcol * DD + kk * 32 + hi * 16);
      v16bf bv = ld_bfrag(Wvb + dcol * DD + kk * 32 + hi * 16);
      ack = WMMA_BF16(afr[kk], bk, ack);
      acv = WMMA_BF16(afr[kk], bv, acv);
    }
#pragma unroll
    for (int r = 0; r < 8; ++r) {
      int row = row0 + hi * 8 + r;
      int b = row / NN, n = row - b * NN;
      float ek = __expf(ack[r]);
      int o = (b * DD + dcol) * NPAD + n;     // transposed store
      ekT[o]  = (__bf16)ek;
      ekvT[o] = (__bf16)(ek * acv[r]);
    }
  }
}

// ---------------------------------------------------------------------------
// Stage 2: q = q1@WqfT + last@WqlT (WMMA); gate = sigmoid(q) (f32).
// Grid: (B*P)/16 = 200 blocks of 128 threads.
// ---------------------------------------------------------------------------
__global__ __launch_bounds__(128) void k_gate(
    const float* __restrict__ xq1, const float* __restrict__ xlast,
    const __bf16* __restrict__ Wqfb, const __bf16* __restrict__ Wqlb,
    float* __restrict__ gate) {
  int tid = threadIdx.x, lane = tid & 31, w = tid >> 5, li = lane & 15, hi = lane >> 4;
  int row0 = blockIdx.x * 16;                 // flat row in [B*P]
  const float* a1 = xq1 + (row0 + li) * DD;
  const float* a2 = xlast + (row0 + li) * DD;
  v16bf f1[4], f2[4];
#pragma unroll
  for (int kk = 0; kk < 4; ++kk) {
    f1[kk] = ld_afrag_f32(a1, kk * 32 + hi * 16);
    f2[kk] = ld_afrag_f32(a2, kk * 32 + hi * 16);
  }
#pragma unroll
  for (int t = 0; t < 2; ++t) {
    int dcol = (w * 2 + t) * 16 + li;
    v8f acc = V8F_ZERO;
#pragma unroll
    for (int kk = 0; kk < 4; ++kk) {
      acc = WMMA_BF16(f1[kk], ld_bfrag(Wqfb + dcol * DD + kk * 32 + hi * 16), acc);
      acc = WMMA_BF16(f2[kk], ld_bfrag(Wqlb + dcol * DD + kk * 32 + hi * 16), acc);
    }
#pragma unroll
    for (int r = 0; r < 8; ++r) {
      int row = row0 + hi * 8 + r;
      gate[row * DD + dcol] = 1.f / (1.f + __expf(-acc[r]));
    }
  }
}

// ---------------------------------------------------------------------------
// Stage 3: num = ebias@(ek*v), den = ebias@ek over K=N (padded 1024);
// aft = gate * num / den, stored bf16. Double-buffered K-loop: fragments for
// step kk+1 are issued before the WMMAs of step kk consume their set.
// Grid: (7 p-tiles, B) blocks of 128 threads. Each wave owns d in [32w,32w+32).
// ---------------------------------------------------------------------------
__global__ __launch_bounds__(128) void k_aft(
    const __bf16* __restrict__ ebias,
    const __bf16* __restrict__ ekT, const __bf16* __restrict__ ekvT,
    const float* __restrict__ gate, __bf16* __restrict__ aftb) {
  int b = blockIdx.y, pt = blockIdx.x;
  int tid = threadIdx.x, lane = tid & 31, w = tid >> 5, li = lane & 15, hi = lane >> 4;
  const __bf16* erow = ebias + (b * PPAD + pt * 16 + li) * NPAD;  // A row
  const __bf16* ekb  = ekT  + b * DD * NPAD;
  const __bf16* ekvb = ekvT + b * DD * NPAD;
  int d0 = w * 32 + li;
  int d1 = w * 32 + 16 + li;
  v8f aN0 = V8F_ZERO, aN1 = V8F_ZERO, aD0 = V8F_ZERO, aD1 = V8F_ZERO;

  v16bf af[2], fn0[2], fn1[2], fd0[2], fd1[2];
  auto load_set = [&](int s, int kk) {
    int ko = kk * 32;
    af[s]  = ld_afrag_bf16(erow, ko + hi * 16);
    fn0[s] = ld_bfrag(ekvb + d0 * NPAD + ko + hi * 16);
    fn1[s] = ld_bfrag(ekvb + d1 * NPAD + ko + hi * 16);
    fd0[s] = ld_bfrag(ekb  + d0 * NPAD + ko + hi * 16);
    fd1[s] = ld_bfrag(ekb  + d1 * NPAD + ko + hi * 16);
  };
  load_set(0, 0);
#pragma unroll 2
  for (int kk = 0; kk < NPAD / 32; ++kk) {
    int cur = kk & 1;
    if (kk + 1 < NPAD / 32) load_set(cur ^ 1, kk + 1);
    aN0 = WMMA_BF16(af[cur], fn0[cur], aN0);
    aN1 = WMMA_BF16(af[cur], fn1[cur], aN1);
    aD0 = WMMA_BF16(af[cur], fd0[cur], aD0);
    aD1 = WMMA_BF16(af[cur], fd1[cur], aD1);
  }
#pragma unroll
  for (int r = 0; r < 8; ++r) {
    int pr = pt * 16 + hi * 8 + r;
    if (pr < PP) {
      int rowg = b * PP + pr;
      float g0 = gate[rowg * DD + d0];
      float g1 = gate[rowg * DD + d1];
      aftb[rowg * DD + d0] = (__bf16)(g0 * aN0[r] / aD0[r]);
      aftb[rowg * DD + d1] = (__bf16)(g1 * aN1[r] / aD1[r]);
    }
  }
}

// ---------------------------------------------------------------------------
// Stage 4: score = aft@nodesT/sqrt(D) - ls*alpha*dist; 10*tanh + mask; exp;
// row-normalize over N (tanh clip bounds scores -> no max pass needed).
// Node B-fragments for tile t+4 prefetched while computing tile t.
// Grid: (7 p-tiles, B) blocks of 128 threads; scores staged in LDS.
// ---------------------------------------------------------------------------
__global__ __launch_bounds__(128) void k_score(
    const float* __restrict__ cur_dist, const float* __restrict__ ninf,
    const float* __restrict__ log_scale, const float* __restrict__ alpha1,
    const __bf16* __restrict__ aftb, const __bf16* __restrict__ nodesb,
    float* __restrict__ out) {
  __shared__ float sS[16 * 1008];             // 63 KB: exp(scores)
  __shared__ float sPart[4][16];              // per-wave row partial sums
  int b = blockIdx.y, pt = blockIdx.x;
  int tid = threadIdx.x, lane = tid & 31, w = tid >> 5, li = lane & 15, hi = lane >> 4;
  float sc = log_scale[0] * alpha1[0];
  const __bf16* arow = aftb + (b * PP + pt * 16 + li) * DD;  // padded alloc
  v16bf afr[4];
#pragma unroll
  for (int kk = 0; kk < 4; ++kk) afr[kk] = ld_afrag_bf16(arow, kk * 32 + hi * 16);
  const __bf16* nb = nodesb + b * NPAD * DD;
  float rsum[8];
#pragma unroll
  for (int r = 0; r < 8; ++r) rsum[r] = 0.f;

  v16bf bfr[2][4];
  auto load_tile = [&](int s, int t) {
    int n = t * 16 + li;
#pragma unroll
    for (int kk = 0; kk < 4; ++kk)
      bfr[s][kk] = ld_bfrag(nb + n * DD + kk * 32 + hi * 16);
  };
  load_tile(0, w);
  int it = 0;
  for (int t = w; t < 63; t += 4, ++it) {     // n-tiles cover n < 1008
    int cur = it & 1;
    if (t + 4 < 63) load_tile(cur ^ 1, t + 4);
    int n = t * 16 + li;
    v8f acc = V8F_ZERO;
#pragma unroll
    for (int kk = 0; kk < 4; ++kk) acc = WMMA_BF16(afr[kk], bfr[cur][kk], acc);
#pragma unroll
    for (int r = 0; r < 8; ++r) {
      int pr = pt * 16 + hi * 8 + r;
      float e = 0.f;
      if (n < NN && pr < PP) {
        int off = (b * PP + pr) * NN + n;
        float s = acc[r] * INV_SQRT_D - sc * cur_dist[off];
        e = __expf(CLIPV * tanhf(s) + ninf[off]);
      }
      sS[(hi * 8 + r) * 1008 + n] = e;
      rsum[r] += e;
    }
  }
  // reduce over the 16-lane half-wave (cols), deterministically across waves
#pragma unroll
  for (int m = 1; m <= 8; m <<= 1)
#pragma unroll
    for (int r = 0; r < 8; ++r) rsum[r] += __shfl_xor(rsum[r], m, 32);
  if (li == 0)
#pragma unroll
    for (int r = 0; r < 8; ++r) sPart[w][hi * 8 + r] = rsum[r];
  __syncthreads();
  for (int i = tid; i < 16 * NN; i += 128) {
    int row = i / NN, n = i - row * NN;
    int pr = pt * 16 + row;
    if (pr < PP) {
      float s = sPart[0][row] + sPart[1][row] + sPart[2][row] + sPart[3][row];
      out[(b * PP + pr) * NN + n] = sS[row * 1008 + n] / s;
    }
  }
}

// ---------------------------------------------------------------------------
// Launcher. Workspace layout (bytes):
//   ekT     [32][128][1024] bf16   8,388,608
//   ekvT    [32][128][1024] bf16   8,388,608
//   nodesb  [32][1024][128] bf16   8,388,608
//   ebias   [32][112][1024] bf16   7,340,032
//   gate    [3200][128]     f32    1,638,400
//   aftb    [3328][128]     bf16     851,968   (rows padded for tile reads)
//   Wkb/Wvb/Wqfb/Wqlb bf16        4 x 32,768
// ---------------------------------------------------------------------------
extern "C" void kernel_launch(void* const* d_in, const int* in_sizes, int n_in,
                              void* d_out, int out_size, void* d_ws, size_t ws_size,
                              hipStream_t stream) {
  const float* enc   = (const float*)d_in[0];
  const float* q1    = (const float*)d_in[1];
  const float* qlast = (const float*)d_in[2];
  const float* cd    = (const float*)d_in[3];
  const float* nm    = (const float*)d_in[4];
  const float* ls    = (const float*)d_in[5];
  const float* Wqf   = (const float*)d_in[6];
  const float* Wql   = (const float*)d_in[7];
  const float* Wk    = (const float*)d_in[8];
  const float* Wv    = (const float*)d_in[9];
  const float* a1    = (const float*)d_in[10];
  const float* aAFT  = (const float*)d_in[11];
  float* out = (float*)d_out;

  char* ws = (char*)d_ws;
  size_t o = 0;
  __bf16* ekT    = (__bf16*)(ws + o); o += (size_t)BB * DD * NPAD * 2;
  __bf16* ekvT   = (__bf16*)(ws + o); o += (size_t)BB * DD * NPAD * 2;
  __bf16* nodesb = (__bf16*)(ws + o); o += (size_t)BB * NPAD * DD * 2;
  __bf16* ebias  = (__bf16*)(ws + o); o += (size_t)BB * PPAD * NPAD * 2;
  float*  gate   = (float*)(ws + o);  o += (size_t)BB * PP * DD * 4;
  __bf16* aftb   = (__bf16*)(ws + o); o += (size_t)3328 * DD * 2;
  __bf16* Wkb    = (__bf16*)(ws + o); o += (size_t)DD * DD * 2;
  __bf16* Wvb    = (__bf16*)(ws + o); o += (size_t)DD * DD * 2;
  __bf16* Wqfb   = (__bf16*)(ws + o); o += (size_t)DD * DD * 2;
  __bf16* Wqlb   = (__bf16*)(ws + o); o += (size_t)DD * DD * 2;

  k_prep<<<384, 256, 0, stream>>>(Wk, Wv, Wqf, Wql, Wkb, Wvb, Wqfb, Wqlb,
                                  ekT, ekvT, nodesb);
  k_ebias<<<(BB * PPAD * NPAD + 255) / 256, 256, 0, stream>>>(cd, nm, ls, aAFT, ebias);
  k_proj_kv<<<(BB * NN) / 16, 128, 0, stream>>>(enc, Wkb, Wvb, ekT, ekvT, nodesb);
  k_gate<<<(BB * PP) / 16, 128, 0, stream>>>(q1, qlast, Wqfb, Wqlb, gate);
  k_aft<<<dim3(7, BB), 128, 0, stream>>>(ebias, ekT, ekvT, gate, aftb);
  k_score<<<dim3(7, BB), 128, 0, stream>>>(cd, nm, ls, a1, aftb, nodesb, out);
}
